// MultiHeadAttention2_72335839199359
// MI455X (gfx1250) — matrix operand order repaired and checked by reading, not compiled, not run on hardware
//
#include <hip/hip_runtime.h>
#include <hip/hip_bf16.h>

#define D_MODEL 1024
#define N_HEADS 16
#define HEAD_DIM 64
#define SEQ 2048
#define BATCH 4
#define MROWS (BATCH * SEQ) // 8192

typedef __bf16 bf16_t;
typedef bf16_t bf16x8  __attribute__((ext_vector_type(8)));
typedef bf16_t bf16x16 __attribute__((ext_vector_type(16)));
typedef float  f32x8   __attribute__((ext_vector_type(8)));

static __device__ __forceinline__ f32x8 wmma_bf(bf16x16 a, bf16x16 b, f32x8 c) {
    // D = A(16x32 bf16) x B(32x16 bf16) + C(16x16 f32)
    return __builtin_amdgcn_wmma_f32_16x16x32_bf16(false, a, false, b, (short)0, c, false, false);
}

// A operand: 16(M) x 32(K) tile from a row-major bf16 matrix (row stride ld).
// Contiguous K order used here (B matches): lanes 0-15 -> row M=lane, K=0..15; lanes 16-31 -> row
// M=lane-16, K=16..31; packed 2 bf16 per VGPR. Two b128 loads per lane.
static __device__ __forceinline__ bf16x16 load_a_frag(const bf16_t* __restrict__ p,
                                                      int ld, int row0, int col0,
                                                      int lq, int half) {
    const bf16_t* r = p + (size_t)(row0 + lq) * ld + col0 + half * 16;
    bf16x8 lo = *(const bf16x8*)(r);        // K = half*16 + 0..7
    bf16x8 hi = *(const bf16x8*)(r + 8);    // K = half*16 + 8..15
    bf16x16 f;
#pragma unroll
    for (int i = 0; i < 8; ++i) { f[i] = lo[i]; f[i + 8] = hi[i]; }
    return f;
}

// B operand: 32(K) x 16(N) tile where column n is contiguous in memory, i.e.
// source is S[n][k] row-major (row stride ld). Lanes 0-15: col N=lane, K=k0..k0+15;
// lanes 16-31: col N=lane-16, K=k0+16..k0+31. Two b128 loads per lane.
static __device__ __forceinline__ bf16x16 load_b_frag(const bf16_t* __restrict__ p,
                                                      int ld, int n0, int k0,
                                                      int lq, int half) {
    const bf16_t* r = p + (size_t)(n0 + lq) * ld + k0 + half * 16;
    bf16x8 lo = *(const bf16x8*)(r);
    bf16x8 hi = *(const bf16x8*)(r + 8);
    bf16x16 f;
#pragma unroll
    for (int i = 0; i < 8; ++i) { f[i] = lo[i]; f[i + 8] = hi[i]; }
    return f;
}

__global__ __launch_bounds__(256) void cvt_f32_bf16(const float* __restrict__ s,
                                                    bf16_t* __restrict__ d, int n) {
    int i = (blockIdx.x * 256 + threadIdx.x) * 4;
    if (i < n) {
        float4 v = *(const float4*)(s + i);
        d[i + 0] = (bf16_t)v.x; d[i + 1] = (bf16_t)v.y;
        d[i + 2] = (bf16_t)v.z; d[i + 3] = (bf16_t)v.w;
    }
}

// C = A(MxK row-major) @ W^T, W stored [n][k] row-major (torch Linear weight).
// Block = 8 waves = 128(M) x 128(N) tile; wave = 32(M) x 64(N) = 8 WMMA accs.
// vmode==1 writes V transposed per head: Vt[b][h][d][s] for the ctx^T GEMM.
__global__ __launch_bounds__(256) void gemm_qkv(const bf16_t* __restrict__ A,
                                                const bf16_t* __restrict__ W,
                                                bf16_t* __restrict__ C, int vmode) {
    const int lane = threadIdx.x & 31, lq = lane & 15, half = lane >> 4;
    const int wv = threadIdx.x >> 5;
    const int m0 = blockIdx.y * 128 + (wv & 3) * 32;
    const int n0 = blockIdx.x * 128 + (wv >> 2) * 64;
    f32x8 zero = {0.f,0.f,0.f,0.f,0.f,0.f,0.f,0.f};
    f32x8 acc[2][4];
#pragma unroll
    for (int i = 0; i < 2; ++i)
#pragma unroll
        for (int j = 0; j < 4; ++j) acc[i][j] = zero;

    for (int k0 = 0; k0 < D_MODEL; k0 += 32) {
        bf16x16 a0 = load_a_frag(A, D_MODEL, m0,      k0, lq, half);
        bf16x16 a1 = load_a_frag(A, D_MODEL, m0 + 16, k0, lq, half);
#pragma unroll
        for (int j = 0; j < 4; ++j) {
            bf16x16 bw = load_b_frag(W, D_MODEL, n0 + j * 16, k0, lq, half);
            acc[0][j] = wmma_bf(a0, bw, acc[0][j]);
            acc[1][j] = wmma_bf(a1, bw, acc[1][j]);
        }
    }
#pragma unroll
    for (int i = 0; i < 2; ++i)
#pragma unroll
        for (int j = 0; j < 4; ++j)
#pragma unroll
            for (int r = 0; r < 8; ++r) {
                int mm = m0 + i * 16 + r + half * 8;   // C layout: M = r + 8*half
                int nn = n0 + j * 16 + lq;             //           N = lane%16
                bf16_t v = (bf16_t)acc[i][j][r];
                if (!vmode) {
                    C[(size_t)mm * D_MODEL + nn] = v;
                } else {
                    int bb = mm >> 11, ss = mm & 2047;
                    int hh = nn >> 6,  dd = nn & 63;
                    C[(((size_t)((bb * N_HEADS + hh) * HEAD_DIM + dd)) << 11) + ss] = v;
                }
            }
}

// Out = Ctx @ Wo^T + bo, fp32 output.
__global__ __launch_bounds__(256) void gemm_out(const bf16_t* __restrict__ A,
                                                const bf16_t* __restrict__ W,
                                                const float* __restrict__ bias,
                                                float* __restrict__ Out) {
    const int lane = threadIdx.x & 31, lq = lane & 15, half = lane >> 4;
    const int wv = threadIdx.x >> 5;
    const int m0 = blockIdx.y * 128 + (wv & 3) * 32;
    const int n0 = blockIdx.x * 128 + (wv >> 2) * 64;
    f32x8 zero = {0.f,0.f,0.f,0.f,0.f,0.f,0.f,0.f};
    f32x8 acc[2][4];
#pragma unroll
    for (int i = 0; i < 2; ++i)
#pragma unroll
        for (int j = 0; j < 4; ++j) acc[i][j] = zero;

    for (int k0 = 0; k0 < D_MODEL; k0 += 32) {
        bf16x16 a0 = load_a_frag(A, D_MODEL, m0,      k0, lq, half);
        bf16x16 a1 = load_a_frag(A, D_MODEL, m0 + 16, k0, lq, half);
#pragma unroll
        for (int j = 0; j < 4; ++j) {
            bf16x16 bw = load_b_frag(W, D_MODEL, n0 + j * 16, k0, lq, half);
            acc[0][j] = wmma_bf(a0, bw, acc[0][j]);
            acc[1][j] = wmma_bf(a1, bw, acc[1][j]);
        }
    }
#pragma unroll
    for (int i = 0; i < 2; ++i)
#pragma unroll
        for (int j = 0; j < 4; ++j)
#pragma unroll
            for (int r = 0; r < 8; ++r) {
                int mm = m0 + i * 16 + r + half * 8;
                int nn = n0 + j * 16 + lq;
                Out[(size_t)mm * D_MODEL + nn] = acc[i][j][r] + bias[nn];
            }
}

// Flash attention: one wave per (b, h, 16-query tile). Computes S^T tiles
// (M=key, N=query) so softmax columns are per-lane; online softmax; P^T is
// rebuilt in the WMMA B-layout with two shfl_xor(16) swaps; ctx accumulated
// transposed: ctx^T = V^T @ P^T.
__global__ __launch_bounds__(256) void attn_flash(const bf16_t* __restrict__ Q,
                                                  const bf16_t* __restrict__ Km,
                                                  const bf16_t* __restrict__ Vt,
                                                  bf16_t* __restrict__ Ctx) {
    const int lane = threadIdx.x & 31;
    const int lq = lane & 15;
    const int half = lane >> 4;
    const int w  = blockIdx.x * 8 + (threadIdx.x >> 5);
    const int qt = w & 127;
    const int h  = (w >> 7) & 15;
    const int b  = w >> 11;
    const int qb = qt * 16;

    const bf16_t* Qb  = Q  + (size_t)b * SEQ * D_MODEL;
    const bf16_t* Kb  = Km + (size_t)b * SEQ * D_MODEL;
    const bf16_t* Vtb = Vt + (size_t)(b * N_HEADS + h) * HEAD_DIM * SEQ;

    // Q^T fragments (B operand), reused across all key blocks: hd=64 -> 2 k-steps
    bf16x16 Bq0 = load_b_frag(Qb, D_MODEL, qb, h * HEAD_DIM,      lq, half);
    bf16x16 Bq1 = load_b_frag(Qb, D_MODEL, qb, h * HEAD_DIM + 32, lq, half);

    const float NEG = -1.0e30f;
    float m_i = NEG, l_i = 0.f;
    f32x8 zero = {0.f,0.f,0.f,0.f,0.f,0.f,0.f,0.f};
    f32x8 acc0 = zero, acc1 = zero, acc2 = zero, acc3 = zero; // ctx^T, M=d 0..63

    const int kend = qb + 16; // causal: keys 0 .. qb+15
    for (int kb = 0; kb < kend; kb += 32) {
        float sc[2][8];
#pragma unroll
        for (int t = 0; t < 2; ++t) {
            int kt = kb + t * 16;
            if (kt < kend) {
                bf16x16 ak0 = load_a_frag(Kb, D_MODEL, kt, h * HEAD_DIM,      lq, half);
                bf16x16 ak1 = load_a_frag(Kb, D_MODEL, kt, h * HEAD_DIM + 32, lq, half);
                f32x8 s = zero;
                s = wmma_bf(ak0, Bq0, s);   // S^T[key][query]
                s = wmma_bf(ak1, Bq1, s);
                bool diag = (kt + 15 > qb); // only diagonal tiles need masking
#pragma unroll
                for (int r = 0; r < 8; ++r) {
                    float v = s[r] * 0.125f; // 1/sqrt(HEAD_DIM)
                    if (diag) {
                        int key = kt + r + half * 8;
                        if (key > qb + lq) v = NEG;
                    }
                    sc[t][r] = v;
                }
            } else {
#pragma unroll
                for (int r = 0; r < 8; ++r) sc[t][r] = NEG;
            }
        }
        // per-query block max: 16 in-lane values + partner half-wave
        float bm = sc[0][0];
#pragma unroll
        for (int r = 1; r < 8; ++r) bm = fmaxf(bm, sc[0][r]);
#pragma unroll
        for (int r = 0; r < 8; ++r) bm = fmaxf(bm, sc[1][r]);
        bm = fmaxf(bm, __shfl_xor(bm, 16, 32));
        float mn = fmaxf(m_i, bm);
        float sf = __expf(m_i - mn);
        m_i = mn;
        float p0[8], p1[8], ps = 0.f;
#pragma unroll
        for (int r = 0; r < 8; ++r) { p0[r] = __expf(sc[0][r] - mn); ps += p0[r]; }
#pragma unroll
        for (int r = 0; r < 8; ++r) { p1[r] = __expf(sc[1][r] - mn); ps += p1[r]; }
        ps += __shfl_xor(ps, 16, 32);
        l_i = l_i * sf + ps;
#pragma unroll
        for (int r = 0; r < 8; ++r) { acc0[r] *= sf; acc1[r] *= sf; acc2[r] *= sf; acc3[r] *= sf; }

        // Rebuild P^T (32 keys x 16 queries) in B-layout:
        // lanes 0-15 need keys kb+0..15 (T0 rows 0..15), lanes 16-31 keys kb+16..31.
        bf16x16 Bp;
#pragma unroll
        for (int r = 0; r < 8; ++r) {
            float o0 = __shfl_xor(p0[r], 16, 32); // T0 rows 8..15 (for lanes<16)
            float o1 = __shfl_xor(p1[r], 16, 32); // T1 rows 0..7  (for lanes>=16)
            Bp[r]     = (bf16_t)(half ? o1 : p0[r]);
            Bp[r + 8] = (bf16_t)(half ? p1[r] : o0);
        }
        // ctx^T += V^T_tile(16d x 32k) @ P^T
        acc0 = wmma_bf(load_a_frag(Vtb, SEQ,  0, kb, lq, half), Bp, acc0);
        acc1 = wmma_bf(load_a_frag(Vtb, SEQ, 16, kb, lq, half), Bp, acc1);
        acc2 = wmma_bf(load_a_frag(Vtb, SEQ, 32, kb, lq, half), Bp, acc2);
        acc3 = wmma_bf(load_a_frag(Vtb, SEQ, 48, kb, lq, half), Bp, acc3);
    }

    float inv = 1.f / l_i;
    bf16_t* out = Ctx + (size_t)(b * SEQ + qb + lq) * D_MODEL + h * HEAD_DIM;
#pragma unroll
    for (int r = 0; r < 8; ++r) {
        int d = r + half * 8; // ctx^T C layout: M=d, N=query(lane)
        out[d]      = (bf16_t)(acc0[r] * inv);
        out[d + 16] = (bf16_t)(acc1[r] * inv);
        out[d + 32] = (bf16_t)(acc2[r] * inv);
        out[d + 48] = (bf16_t)(acc3[r] * inv);
    }
}

extern "C" void kernel_launch(void* const* d_in, const int* in_sizes, int n_in,
                              void* d_out, int out_size, void* d_ws, size_t ws_size,
                              hipStream_t stream) {
    (void)in_sizes; (void)n_in; (void)out_size; (void)ws_size;
    const float* x  = (const float*)d_in[0];
    const float* Wq = (const float*)d_in[1];
    const float* Wk = (const float*)d_in[2];
    const float* Wv = (const float*)d_in[3];
    const float* Wo = (const float*)d_in[4];
    const float* bo = (const float*)d_in[5];

    char* ws = (char*)d_ws;
    const size_t MB = (size_t)1 << 20;
    bf16_t* Xbf = (bf16_t*)(ws +  0 * MB);  // 16 MiB, reused as Ctx after QKV
    bf16_t* Qbf = (bf16_t*)(ws + 16 * MB);  // 16 MiB
    bf16_t* Kbf = (bf16_t*)(ws + 32 * MB);  // 16 MiB
    bf16_t* Vtb = (bf16_t*)(ws + 48 * MB);  // 16 MiB, [b][h][d][s]
    bf16_t* Wqb = (bf16_t*)(ws + 64 * MB);  // 2 MiB each
    bf16_t* Wkb = (bf16_t*)(ws + 66 * MB);
    bf16_t* Wvb = (bf16_t*)(ws + 68 * MB);
    bf16_t* Wob = (bf16_t*)(ws + 70 * MB);
    bf16_t* Ctxb = Xbf;                     // X dead after QKV GEMMs

    cvt_f32_bf16<<<MROWS * D_MODEL / 1024, 256, 0, stream>>>(x, Xbf, MROWS * D_MODEL);
    cvt_f32_bf16<<<D_MODEL * D_MODEL / 1024, 256, 0, stream>>>(Wq, Wqb, D_MODEL * D_MODEL);
    cvt_f32_bf16<<<D_MODEL * D_MODEL / 1024, 256, 0, stream>>>(Wk, Wkb, D_MODEL * D_MODEL);
    cvt_f32_bf16<<<D_MODEL * D_MODEL / 1024, 256, 0, stream>>>(Wv, Wvb, D_MODEL * D_MODEL);
    cvt_f32_bf16<<<D_MODEL * D_MODEL / 1024, 256, 0, stream>>>(Wo, Wob, D_MODEL * D_MODEL);

    dim3 gg(D_MODEL / 128, MROWS / 128);   // (8, 64) blocks of 8 waves
    gemm_qkv<<<gg, 256, 0, stream>>>(Xbf, Wqb, Qbf, 0);
    gemm_qkv<<<gg, 256, 0, stream>>>(Xbf, Wkb, Kbf, 0);
    gemm_qkv<<<gg, 256, 0, stream>>>(Xbf, Wvb, Vtb, 1);

    attn_flash<<<(BATCH * N_HEADS * (SEQ / 16)) / 8, 256, 0, stream>>>(Qbf, Kbf, Vtb, Ctxb);

    gemm_out<<<gg, 256, 0, stream>>>(Ctxb, Wob, bo, (float*)d_out);
}
